// DeformableTransformerEncoderLayer_16406775071311
// MI455X (gfx1250) — matrix operand order repaired and checked by reading, not compiled, not run on hardware
//
#include <hip/hip_runtime.h>
#include <hip/hip_bf16.h>
#include <math.h>

// ---------------------------------------------------------------------------
// Problem constants (from reference)
// ---------------------------------------------------------------------------
#define LQ       20197           // sum of H*W over levels
#define BATCH    2
#define MROWS    (BATCH * LQ)    // 40394
#define DMODEL   256
#define DFFN     1024
#define NHEADS   8
#define NLEV     4
#define NPTS     4

typedef __attribute__((ext_vector_type(16))) _Float16 v16h;
typedef __attribute__((ext_vector_type(8)))  float    v8f;

// ---------------------------------------------------------------------------
// Pack W[K,N] (f32, row-major) into WMMA B-fragment order (f16):
//   For K-chunk c (32 rows), col-tile j (16 cols), lane l (0..31):
//     n = j*16 + (l&15),  kbase = c*32 + (l>=16 ? 8 : 0)
//     P[((c*(N/16)+j)*32 + l)*16 + h] = f16( W[(kbase+h+(h>=8?8:0))*N + n] ),  h=0..15
// Each lane's fragment is 32 contiguous bytes -> one v16h load in the GEMM.
// ---------------------------------------------------------------------------
__global__ __launch_bounds__(256) void pack_weights(
    const float* __restrict__ W, _Float16* __restrict__ P, int K, int N)
{
    const int tid   = blockIdx.x * 256 + threadIdx.x;
    const int NT    = N >> 4;
    const int total = (K >> 5) * NT * 32;
    if (tid >= total) return;

    const int lane = tid & 31;
    const int j    = (tid >> 5) % NT;
    const int c    = (tid >> 5) / NT;
    const int n    = j * 16 + (lane & 15);
    const int kb   = c * 32 + ((lane >> 4) << 3);

    union { v16h v; _Float16 e[16]; } u;
    #pragma unroll
    for (int h = 0; h < 16; ++h)
        u.e[h] = (_Float16)W[(size_t)(kb + h + ((h >> 3) << 3)) * N + n];
    *(v16h*)(P + (size_t)tid * 16) = u.v;
}

// ---------------------------------------------------------------------------
// LDS-free WMMA GEMM: C[M,N] = (A (+A2 optional))[M,K] @ W[K,N] + bias[N]
// A: f32 row-major (fragments loaded directly: 4x b128 + cvt to f16)
// W: pre-packed f16 fragments (1x 32B v16h load per tile per chunk)
// Block = 256 threads = 8 waves; wave w owns rows [m0+16w, m0+16w+16) x 64 cols
// -> 4 v_wmma_f32_16x16x32_f16 per K-chunk per wave, f32 accumulation.
// ---------------------------------------------------------------------------
__global__ __launch_bounds__(256) void gemm_wmma_packed(
    const float* __restrict__ A, const float* __restrict__ A2,
    const _Float16* __restrict__ Wp, const float* __restrict__ bias,
    float* __restrict__ C, int M, int N, int K, int relu)
{
    const int lane = threadIdx.x & 31;
    const int w    = threadIdx.x >> 5;              // 0..7 : 16-row strip
    const int m0   = blockIdx.y * 128 + w * 16;
    const int n0   = blockIdx.x * 64;
    const int NT   = N >> 4;
    const int jt   = n0 >> 4;                       // first of 4 col-tiles

    // A fragment addressing (16-bit A 16x32 ISA layout):
    // lane row = lane&15 ; k-halves {kb..kb+7, kb+16..kb+23}, kb = (lane>=16)?8:0
    int grow = m0 + (lane & 15);
    if (grow >= M) grow = M - 1;                    // clamp; stores guarded below
    const int   kb  = (lane >> 4) << 3;
    const float* pa  = A + (size_t)grow * K + kb;
    const float* pa2 = A2 ? (A2 + (size_t)grow * K + kb) : nullptr;

    v8f c0 = {}, c1 = {}, c2 = {}, c3 = {};

    #pragma unroll 2
    for (int k0 = 0; k0 < K; k0 += 32) {
        // ---- A fragment: two contiguous 8-float runs -> 4 x b128 + cvt ----
        union { v16h v; _Float16 e[16]; } af;
        {
            const float* p = pa + k0;
            float4 f0 = *(const float4*)(p + 0);
            float4 f1 = *(const float4*)(p + 4);
            float4 f2 = *(const float4*)(p + 16);
            float4 f3 = *(const float4*)(p + 20);
            if (pa2) {
                const float* q = pa2 + k0;
                float4 g0 = *(const float4*)(q + 0);
                float4 g1 = *(const float4*)(q + 4);
                float4 g2 = *(const float4*)(q + 16);
                float4 g3 = *(const float4*)(q + 20);
                f0.x += g0.x; f0.y += g0.y; f0.z += g0.z; f0.w += g0.w;
                f1.x += g1.x; f1.y += g1.y; f1.z += g1.z; f1.w += g1.w;
                f2.x += g2.x; f2.y += g2.y; f2.z += g2.z; f2.w += g2.w;
                f3.x += g3.x; f3.y += g3.y; f3.z += g3.z; f3.w += g3.w;
            }
            af.e[0]  = (_Float16)f0.x; af.e[1]  = (_Float16)f0.y;
            af.e[2]  = (_Float16)f0.z; af.e[3]  = (_Float16)f0.w;
            af.e[4]  = (_Float16)f1.x; af.e[5]  = (_Float16)f1.y;
            af.e[6]  = (_Float16)f1.z; af.e[7]  = (_Float16)f1.w;
            af.e[8]  = (_Float16)f2.x; af.e[9]  = (_Float16)f2.y;
            af.e[10] = (_Float16)f2.z; af.e[11] = (_Float16)f2.w;
            af.e[12] = (_Float16)f3.x; af.e[13] = (_Float16)f3.y;
            af.e[14] = (_Float16)f3.z; af.e[15] = (_Float16)f3.w;
        }

        // ---- B fragments: one 32B v16h load per 16-col tile ----
        const _Float16* bp =
            Wp + ((size_t)(k0 >> 5) * NT + jt) * 512 + lane * 16;
        v16h b0 = *(const v16h*)(bp);
        v16h b1 = *(const v16h*)(bp + 512);
        v16h b2 = *(const v16h*)(bp + 1024);
        v16h b3 = *(const v16h*)(bp + 1536);

        c0 = __builtin_amdgcn_wmma_f32_16x16x32_f16(false, af.v, false, b0,
                                                    (short)0, c0, false, false);
        c1 = __builtin_amdgcn_wmma_f32_16x16x32_f16(false, af.v, false, b1,
                                                    (short)0, c1, false, false);
        c2 = __builtin_amdgcn_wmma_f32_16x16x32_f16(false, af.v, false, b2,
                                                    (short)0, c2, false, false);
        c3 = __builtin_amdgcn_wmma_f32_16x16x32_f16(false, af.v, false, b3,
                                                    (short)0, c3, false, false);
    }

    // ---- epilogue: v8f element r -> M = r + (lane>=16 ? 8 : 0), N = lane&15
    const int mbase = m0 + ((lane >> 4) << 3);
    const int nb    = n0 + (lane & 15);
    const float bia0 = bias[nb +  0];
    const float bia1 = bias[nb + 16];
    const float bia2 = bias[nb + 32];
    const float bia3 = bias[nb + 48];
    #pragma unroll
    for (int r = 0; r < 8; ++r) {
        const int gm = mbase + r;
        if (gm < M) {
            float v0 = c0[r] + bia0;
            float v1 = c1[r] + bia1;
            float v2 = c2[r] + bia2;
            float v3 = c3[r] + bia3;
            if (relu) {
                v0 = fmaxf(v0, 0.0f); v1 = fmaxf(v1, 0.0f);
                v2 = fmaxf(v2, 0.0f); v3 = fmaxf(v3, 0.0f);
            }
            float* pr = C + (size_t)gm * N + nb;
            pr[0]  = v0;
            pr[16] = v1;
            pr[32] = v2;
            pr[48] = v3;
        }
    }
}

// ---------------------------------------------------------------------------
// MS-deformable-attention sampling: one wave32 per (b,q,head); lane = channel.
// ---------------------------------------------------------------------------
__global__ __launch_bounds__(256) void msda_sample(
    const float* __restrict__ value,   // [B, LQ, 256]
    const float* __restrict__ off,     // [B, LQ, 256]
    const float* __restrict__ attnlog, // [B, LQ, 128]
    const float* __restrict__ ref,     // [B, LQ, 4, 2]
    float* __restrict__ out,           // [B, LQ, 256]
    int total)                          // B*LQ*NHEADS
{
    const int gw = blockIdx.x * 8 + (threadIdx.x >> 5);
    if (gw >= total) return;
    const int lane = threadIdx.x & 31;
    const int h  = gw & 7;
    const int bq = gw >> 3;
    const int b  = bq / LQ;

    const int LH[NLEV] = {100, 50, 25, 13};
    const int LW[NLEV] = {152, 76, 38, 19};
    const int LS[NLEV] = {0, 15200, 19000, 19950};

    // softmax over the 16 (level,point) logits for this head (wave-uniform)
    const float* al = attnlog + (size_t)bq * 128 + h * 16;
    float wgt[16];
    float mx = -1e30f;
    #pragma unroll
    for (int i = 0; i < 16; ++i) { wgt[i] = al[i]; mx = fmaxf(mx, wgt[i]); }
    float s = 0.0f;
    #pragma unroll
    for (int i = 0; i < 16; ++i) { wgt[i] = __expf(wgt[i] - mx); s += wgt[i]; }
    const float inv = 1.0f / s;

    const float* offp  = off + (size_t)bq * 256 + h * 32;
    const float* refp  = ref + (size_t)bq * 8;
    const float* vbase = value + (size_t)b * LQ * 256 + h * 32 + lane;

    float acc = 0.0f;
    #pragma unroll
    for (int l = 0; l < NLEV; ++l) {
        const float rx = refp[l * 2 + 0];
        const float ry = refp[l * 2 + 1];
        const float Wf = (float)LW[l];
        const float Hf = (float)LH[l];
        const float* vl = vbase + (size_t)LS[l] * 256;
        #pragma unroll
        for (int p = 0; p < NPTS; ++p) {
            const float ox = offp[l * 8 + p * 2 + 0];
            const float oy = offp[l * 8 + p * 2 + 1];
            const float x  = (rx + ox / Wf) * Wf - 0.5f;
            const float y  = (ry + oy / Hf) * Hf - 0.5f;
            const float x0 = floorf(x);
            const float y0 = floorf(y);
            const float aw = wgt[l * 4 + p] * inv;
            #pragma unroll
            for (int dy = 0; dy < 2; ++dy) {
                #pragma unroll
                for (int dx = 0; dx < 2; ++dx) {
                    const float xi = x0 + (float)dx;
                    const float yi = y0 + (float)dy;
                    if (xi >= 0.0f && xi < Wf && yi >= 0.0f && yi < Hf) {
                        const float wb = (1.0f - fabsf(x - xi)) * (1.0f - fabsf(y - yi));
                        const int idx = (int)yi * LW[l] + (int)xi;
                        acc += aw * wb * vl[(size_t)idx * 256];
                    }
                }
            }
        }
    }
    out[(size_t)bq * 256 + h * 32 + lane] = acc;
}

// ---------------------------------------------------------------------------
// LayerNorm(a + b) * g + be : one wave32 per 256-channel row, shuffle reduce.
// ---------------------------------------------------------------------------
__global__ __launch_bounds__(256) void ln_residual(
    const float* __restrict__ a, const float* __restrict__ bres,
    const float* __restrict__ g, const float* __restrict__ be,
    float* __restrict__ out, int rows)
{
    const int row = blockIdx.x * 8 + (threadIdx.x >> 5);
    if (row >= rows) return;
    const int lane = threadIdx.x & 31;
    const float* pa = a    + (size_t)row * DMODEL;
    const float* pb = bres + (size_t)row * DMODEL;

    float v[8];
    float s = 0.0f, s2 = 0.0f;
    #pragma unroll
    for (int j = 0; j < 8; ++j) {
        const int c = lane + j * 32;
        const float t = pa[c] + pb[c];
        v[j] = t; s += t; s2 += t * t;
    }
    #pragma unroll
    for (int m = 16; m > 0; m >>= 1) {
        s  += __shfl_xor(s,  m, 32);
        s2 += __shfl_xor(s2, m, 32);
    }
    const float mean = s * (1.0f / 256.0f);
    const float var  = s2 * (1.0f / 256.0f) - mean * mean;
    const float rstd = rsqrtf(var + 1e-5f);
    #pragma unroll
    for (int j = 0; j < 8; ++j) {
        const int c = lane + j * 32;
        out[(size_t)row * DMODEL + c] = (v[j] - mean) * rstd * g[c] + be[c];
    }
}

// ---------------------------------------------------------------------------
// Launch sequence
// ---------------------------------------------------------------------------
extern "C" void kernel_launch(void* const* d_in, const int* in_sizes, int n_in,
                              void* d_out, int out_size, void* d_ws, size_t ws_size,
                              hipStream_t stream)
{
    (void)in_sizes; (void)n_in; (void)out_size; (void)ws_size;

    const float* src    = (const float*)d_in[0];
    const float* pos    = (const float*)d_in[1];
    const float* refpts = (const float*)d_in[2];
    const float* W_off  = (const float*)d_in[5];
    const float* b_off  = (const float*)d_in[6];
    const float* W_attn = (const float*)d_in[7];
    const float* b_attn = (const float*)d_in[8];
    const float* W_val  = (const float*)d_in[9];
    const float* b_val  = (const float*)d_in[10];
    const float* W_out  = (const float*)d_in[11];
    const float* b_out  = (const float*)d_in[12];
    const float* W1     = (const float*)d_in[13];
    const float* b1     = (const float*)d_in[14];
    const float* W2     = (const float*)d_in[15];
    const float* b2     = (const float*)d_in[16];
    const float* g1     = (const float*)d_in[17];
    const float* be1    = (const float*)d_in[18];
    const float* g2     = (const float*)d_in[19];
    const float* be2    = (const float*)d_in[20];

    const int M = MROWS;

    // workspace carve-up
    float* value   = (float*)d_ws;                       // M*256
    float* offbuf  = value   + (size_t)M * 256;          // M*256
    float* attnbuf = offbuf  + (size_t)M * 256;          // M*128
    float* sampled = attnbuf + (size_t)M * 128;          // M*256
    float* src2    = sampled + (size_t)M * 256;          // M*256
    float* xbuf    = src2    + (size_t)M * 256;          // M*256
    float* hbuf    = xbuf    + (size_t)M * 256;          // M*1024
    float* ffnbuf  = hbuf    + (size_t)M * 1024;         // M*256

    _Float16* WpVal  = (_Float16*)(ffnbuf + (size_t)M * 256);
    _Float16* WpOff  = WpVal  + (size_t)256 * 256;
    _Float16* WpAttn = WpOff  + (size_t)256 * 256;
    _Float16* WpOut  = WpAttn + (size_t)256 * 128;
    _Float16* Wp1    = WpOut  + (size_t)256 * 256;
    _Float16* Wp2    = Wp1    + (size_t)256 * 1024;

    const dim3 blk(256);
    const int mb = (M + 127) / 128;

    // ---- pack all weights into WMMA fragment layout (f16) ----
    pack_weights<<<dim3((256 * 256 / 16 + 255) / 256), blk, 0, stream>>>(W_val,  WpVal,  256, 256);
    pack_weights<<<dim3((256 * 256 / 16 + 255) / 256), blk, 0, stream>>>(W_off,  WpOff,  256, 256);
    pack_weights<<<dim3((256 * 128 / 16 + 255) / 256), blk, 0, stream>>>(W_attn, WpAttn, 256, 128);
    pack_weights<<<dim3((256 * 256 / 16 + 255) / 256), blk, 0, stream>>>(W_out,  WpOut,  256, 256);
    pack_weights<<<dim3((256 * 1024 / 16 + 255) / 256), blk, 0, stream>>>(W1,    Wp1,    256, 1024);
    pack_weights<<<dim3((1024 * 256 / 16 + 255) / 256), blk, 0, stream>>>(W2,    Wp2,    1024, 256);

    // 1) value = src @ W_val + b_val
    gemm_wmma_packed<<<dim3(DMODEL / 64, mb), blk, 0, stream>>>(
        src, nullptr, WpVal, b_val, value, M, DMODEL, DMODEL, 0);
    // 2) off = (src+pos) @ W_off + b_off
    gemm_wmma_packed<<<dim3(DMODEL / 64, mb), blk, 0, stream>>>(
        src, pos, WpOff, b_off, offbuf, M, DMODEL, DMODEL, 0);
    // 3) attn logits = (src+pos) @ W_attn + b_attn
    gemm_wmma_packed<<<dim3(128 / 64, mb), blk, 0, stream>>>(
        src, pos, WpAttn, b_attn, attnbuf, M, 128, DMODEL, 0);

    // 4) deformable sampling + softmax + weighted sum
    const int total_waves = M * NHEADS;
    msda_sample<<<dim3((total_waves + 7) / 8), blk, 0, stream>>>(
        value, offbuf, attnbuf, refpts, sampled, total_waves);

    // 5) src2 = sampled @ W_out + b_out
    gemm_wmma_packed<<<dim3(DMODEL / 64, mb), blk, 0, stream>>>(
        sampled, nullptr, WpOut, b_out, src2, M, DMODEL, DMODEL, 0);

    // 6) x = LN(src + src2)
    ln_residual<<<dim3((M + 7) / 8), blk, 0, stream>>>(src, src2, g1, be1, xbuf, M);

    // 7) h = relu(x @ W1 + b1)
    gemm_wmma_packed<<<dim3(DFFN / 64, mb), blk, 0, stream>>>(
        xbuf, nullptr, Wp1, b1, hbuf, M, DFFN, DMODEL, 1);

    // 8) ffn = h @ W2 + b2
    gemm_wmma_packed<<<dim3(DMODEL / 64, mb), blk, 0, stream>>>(
        hbuf, nullptr, Wp2, b2, ffnbuf, M, DMODEL, DFFN, 0);

    // 9) out = LN(x + ffn)
    ln_residual<<<dim3((M + 7) / 8), blk, 0, stream>>>(
        xbuf, ffnbuf, g2, be2, (float*)d_out, M);
}
